// TensorRf_34866544509605
// MI455X (gfx1250) — matrix operand order repaired and checked by reading, not compiled, not run on hardware
//
#include <hip/hip_runtime.h>
#include <hip/hip_bf16.h>

#define RADIUSF 1.5f
#define RESI 300
#define NSIG 16
#define NRGB 48
#define NSAMP 259
#define STEPF 0.020066889632107024f
#define THRESH 0.0001f
#define NRAYS 2048
#define PHW 90000

typedef _Float16 half16 __attribute__((ext_vector_type(16)));
typedef _Float16 half4  __attribute__((ext_vector_type(4)));
typedef float    v8f    __attribute__((ext_vector_type(8)));

union HQ { half16 h; uint4 q[2]; };

// ---- gather helpers: TR=true -> channel-last repacked layout (float4 taps) ----
template<bool TR>
__device__ __forceinline__ float4 ldc4(const float* __restrict__ base, int pt, int c, int C, int HW) {
    if (TR) {
        return *(const float4*)(base + (size_t)pt * C + c);
    } else {
        float4 r;
        r.x = base[(size_t)(c + 0) * HW + pt];
        r.y = base[(size_t)(c + 1) * HW + pt];
        r.z = base[(size_t)(c + 2) * HW + pt];
        r.w = base[(size_t)(c + 3) * HW + pt];
        return r;
    }
}

struct Bil { int t00; float w00, w01, w10, w11; };
struct Lin { int r0; float w0, w1; };

__device__ __forceinline__ Bil mk_bil(float ax, float ay) {
    float x = (ax + 1.f) * 0.5f * (float)(RESI - 1);
    float y = (ay + 1.f) * 0.5f * (float)(RESI - 1);
    float xf = fminf(fmaxf(floorf(x), 0.f), (float)(RESI - 2));
    float yf = fminf(fmaxf(floorf(y), 0.f), (float)(RESI - 2));
    float wx = fminf(fmaxf(x - xf, 0.f), 1.f);
    float wy = fminf(fmaxf(y - yf, 0.f), 1.f);
    Bil b;
    b.t00 = (int)yf * RESI + (int)xf;
    float ux = 1.f - wx, uy = 1.f - wy;
    b.w00 = uy * ux; b.w01 = uy * wx; b.w10 = wy * ux; b.w11 = wy * wx;
    return b;
}

__device__ __forceinline__ Lin mk_lin(float az) {
    float y = (az + 1.f) * 0.5f * (float)(RESI - 1);
    float yf = fminf(fmaxf(floorf(y), 0.f), (float)(RESI - 2));
    float wy = fminf(fmaxf(y - yf, 0.f), 1.f);
    Lin l; l.r0 = (int)yf; l.w1 = wy; l.w0 = 1.f - wy;
    return l;
}

template<bool TR>
__device__ __forceinline__ float4 bil4(const float* __restrict__ pl, const Bil& b, int c, int C) {
    float4 v00 = ldc4<TR>(pl, b.t00,            c, C, PHW);
    float4 v01 = ldc4<TR>(pl, b.t00 + 1,        c, C, PHW);
    float4 v10 = ldc4<TR>(pl, b.t00 + RESI,     c, C, PHW);
    float4 v11 = ldc4<TR>(pl, b.t00 + RESI + 1, c, C, PHW);
    float4 r;
    r.x = v00.x * b.w00 + v01.x * b.w01 + v10.x * b.w10 + v11.x * b.w11;
    r.y = v00.y * b.w00 + v01.y * b.w01 + v10.y * b.w10 + v11.y * b.w11;
    r.z = v00.z * b.w00 + v01.z * b.w01 + v10.z * b.w10 + v11.z * b.w11;
    r.w = v00.w * b.w00 + v01.w * b.w01 + v10.w * b.w10 + v11.w * b.w11;
    return r;
}

template<bool TR>
__device__ __forceinline__ float4 lin4(const float* __restrict__ ln, const Lin& l, int c, int C) {
    float4 a = ldc4<TR>(ln, l.r0,     c, C, RESI);
    float4 b = ldc4<TR>(ln, l.r0 + 1, c, C, RESI);
    float4 r;
    r.x = a.x * l.w0 + b.x * l.w1;
    r.y = a.y * l.w0 + b.y * l.w1;
    r.z = a.z * l.w0 + b.z * l.w1;
    r.w = a.w * l.w0 + b.w * l.w1;
    return r;
}

// ---- channel-last repack: (3,C,HW) -> (3,HW,C) ----
template<int C>
__global__ void __launch_bounds__(256, 1) k_tp(const float* __restrict__ in, float* __restrict__ out, int HW) {
    __shared__ float tile[64 * C];
    int p = blockIdx.y;
    int y0 = blockIdx.x * 64;
    const float* ip = in + (size_t)p * C * HW;
    float* op = out + (size_t)p * HW * C;
    for (int i = threadIdx.x; i < 64 * C; i += 256) {
        int c = i >> 6, j = i & 63;
        int yx = y0 + j;
        tile[j * C + c] = (yx < HW) ? ip[(size_t)c * HW + yx] : 0.f;
    }
    __syncthreads();
    for (int i = threadIdx.x; i < 64 * C; i += 256) {
        int j = i / C, c = i - j * C;
        int yx = y0 + j;
        if (yx < HW) op[(size_t)yx * C + c] = tile[j * C + c];
    }
}

// ---- main render kernel: 1 wave (32 lanes) per ray, 4 rays per block ----
template<bool TR>
__global__ void __launch_bounds__(128, 1) k_render(
    const float* __restrict__ rays_o, const float* __restrict__ rays_d,
    const float* __restrict__ dP, const float* __restrict__ dL,
    const float* __restrict__ rP, const float* __restrict__ rL,
    const float* __restrict__ Wb, float* __restrict__ out)
{
    __shared__ __align__(16) _Float16 Wh[32][160];      // basis_W^T padded to 32x160 (f16)
    __shared__ __align__(16) _Float16 At[4][32][160];   // per-wave A tiles (32 samples x 160 K)
    __shared__ __align__(16) float    Cout[4][32][28];  // per-wave WMMA outputs (col 27 = pad sink)

    const int tid = threadIdx.x;
    for (int i = tid; i < 32 * 160; i += 128) {
        int n = i / 160, k = i - n * 160;
        float v = (n < 27 && k < 144) ? Wb[n * 144 + k] : 0.f;
        Wh[n][k] = (_Float16)v;
    }
    __syncthreads();

    const int wv   = tid >> 5;
    const int lane = tid & 31;
    const int lr   = lane & 15;
    const int hs   = lane >> 4;
    const int ray  = blockIdx.x * 4 + wv;

    // Hoist all B fragments (loop-invariant, lane-dependent only) into registers:
    // 2 N-tiles x 5 K-chunks x 16 halfs = 80 VGPRs.
    half16 Bf[2][5];
    #pragma unroll
    for (int nt = 0; nt < 2; ++nt) {
        #pragma unroll
        for (int kc = 0; kc < 5; ++kc) {
            HQ B;
            int kb = kc * 32 + hs * 8;
            B.q[0] = *(const uint4*)&Wh[nt * 16 + lr][kb];
            B.q[1] = *(const uint4*)&Wh[nt * 16 + lr][kb + 16];
            Bf[nt][kc] = B.h;
        }
    }

    const float ox = rays_o[ray * 3 + 0], oy = rays_o[ray * 3 + 1], oz = rays_o[ray * 3 + 2];
    const float dx = rays_d[ray * 3 + 0], dy = rays_d[ray * 3 + 1], dz = rays_d[ray * 3 + 2];

    float tmin;
    {
        float vx = (dx == 0.f) ? 1e-6f : dx;
        float vy = (dy == 0.f) ? 1e-6f : dy;
        float vz = (dz == 0.f) ? 1e-6f : dz;
        float tx = fminf((RADIUSF - ox) / vx, (-RADIUSF - ox) / vx);
        float ty = fminf((RADIUSF - oy) / vy, (-RADIUSF - oy) / vy);
        float tz = fminf((RADIUSF - oz) / vz, (-RADIUSF - oz) / vz);
        tmin = fmaxf(fmaxf(tx, ty), tz);
    }
    const float dnorm = sqrtf(dx * dx + dy * dy + dz * dz);
    const float dstep = STEPF * dnorm;

    float sh[9];
    {
        const float C0 = 0.28209479177387814f, C1 = 0.4886025119029199f;
        const float C2a = 1.0925484305920792f, C2c = 0.31539156525252005f, C2e = 0.5462742152960396f;
        sh[0] = C0;
        sh[1] = -C1 * dy;
        sh[2] =  C1 * dz;
        sh[3] = -C1 * dx;
        sh[4] =  C2a * dx * dy;
        sh[5] = -C2a * dy * dz;
        sh[6] =  C2c * (2.f * dz * dz - dx * dx - dy * dy);
        sh[7] = -C2a * dx * dz;
        sh[8] =  C2e * (dx * dx - dy * dy);
    }

    float carry = 1.f;
    float racc = 0.f, gacc = 0.f, bacc = 0.f, wsum = 0.f, dsum = 0.f;

    for (int base = 0; base < NSAMP; base += 32) {
        int idx = base + lane;
        float t = tmin + STEPF * (float)idx;
        float px = fmaf(dx, t, ox);
        float py = fmaf(dy, t, oy);
        float pz = fmaf(dz, t, oz);
        bool inb = (idx < NSAMP) &&
                   px >= -RADIUSF && px <= RADIUSF &&
                   py >= -RADIUSF && py <= RADIUSF &&
                   pz >= -RADIUSF && pz <= RADIUSF;
        float nx = (px + RADIUSF) * (1.f / (2.f * RADIUSF)) - 1.f;
        float ny = (py + RADIUSF) * (1.f / (2.f * RADIUSF)) - 1.f;
        float nz = (pz + RADIUSF) * (1.f / (2.f * RADIUSF)) - 1.f;
        float pax[3] = { nx, nx, ny };
        float pay[3] = { ny, nz, nz };
        float paz[3] = { nz, ny, nx };

        // -------- density -> sigma --------
        float sigma = 0.f;
        if (inb) {
            #pragma unroll
            for (int p = 0; p < 3; ++p) {
                Bil bl = mk_bil(pax[p], pay[p]);
                Lin li = mk_lin(paz[p]);
                const float* pl = dP + (size_t)p * NSIG * PHW;
                const float* ln = dL + (size_t)p * NSIG * RESI;
                #pragma unroll
                for (int c = 0; c < NSIG; c += 4) {
                    float4 pb = bil4<TR>(pl, bl, c, NSIG);
                    float4 lv = lin4<TR>(ln, li, c, NSIG);
                    sigma += pb.x * lv.x + pb.y * lv.y + pb.z * lv.z + pb.w * lv.w;
                }
            }
            sigma = fmaxf(sigma, 0.f);
        }
        float alpha = inb ? (1.f - expf(-sigma * dstep)) : 0.f;

        // -------- wave32 prefix-product transmittance scan --------
        float tr = 1.f - alpha + 1e-10f;
        float pprod = tr;
        #pragma unroll
        for (int o2 = 1; o2 < 32; o2 <<= 1) {
            float v = __shfl_up(pprod, o2);
            pprod = (lane >= o2) ? pprod * v : pprod;
        }
        float pex = __shfl_up(pprod, 1);
        if (lane == 0) pex = 1.f;
        float w = alpha * carry * pex;
        carry *= __shfl(pprod, 31);

        dsum += w * t;
        wsum += w;

        bool active = inb && (w > THRESH);
        float cr = 0.5f, cg = 0.5f, cb = 0.5f;

        if (__any(active)) {
            // -------- build 32x160 f16 feature tile (A) --------
            _Float16* arow = &At[wv][lane][0];
            if (active) {
                #pragma unroll
                for (int p = 0; p < 3; ++p) {
                    Bil bl = mk_bil(pax[p], pay[p]);
                    Lin li = mk_lin(paz[p]);
                    const float* pl = rP + (size_t)p * NRGB * PHW;
                    const float* ln = rL + (size_t)p * NRGB * RESI;
                    #pragma unroll
                    for (int c = 0; c < NRGB; c += 4) {
                        float4 pb = bil4<TR>(pl, bl, c, NRGB);
                        float4 lv = lin4<TR>(ln, li, c, NRGB);
                        half4 h;
                        h.x = (_Float16)(pb.x * lv.x);
                        h.y = (_Float16)(pb.y * lv.y);
                        h.z = (_Float16)(pb.z * lv.z);
                        h.w = (_Float16)(pb.w * lv.w);
                        *(half4*)&arow[p * NRGB + c] = h;
                    }
                }
            } else {
                half4 z; z.x = z.y = z.z = z.w = (_Float16)0.f;
                #pragma unroll
                for (int k = 0; k < 144; k += 4) *(half4*)&arow[k] = z;
            }
            {
                half4 z; z.x = z.y = z.z = z.w = (_Float16)0.f;
                *(half4*)&arow[144] = z; *(half4*)&arow[148] = z;
                *(half4*)&arow[152] = z; *(half4*)&arow[156] = z;
            }

            // -------- WMMA: (32x144 feat) x (144x27 W^T) -> 32x27 --------
            const int col0 = lr;
            const int col1 = (lr + 16 < 27) ? (lr + 16) : 27;  // pad-sink column, unconditional store
            #pragma unroll
            for (int m = 0; m < 2; ++m) {
                v8f c0 = {0.f,0.f,0.f,0.f,0.f,0.f,0.f,0.f};
                v8f c1 = {0.f,0.f,0.f,0.f,0.f,0.f,0.f,0.f};
                const _Float16* ar = &At[wv][m * 16 + lr][0];
                #pragma unroll
                for (int kc = 0; kc < 5; ++kc) {
                    int k0 = kc * 32;
                    HQ A;
                    A.q[0] = *(const uint4*)&ar[k0 + hs * 8];
                    A.q[1] = *(const uint4*)&ar[k0 + hs * 8 + 16];
                    c0 = __builtin_amdgcn_wmma_f32_16x16x32_f16(false, A.h, false, Bf[0][kc], (short)0, c0, false, false);
                    c1 = __builtin_amdgcn_wmma_f32_16x16x32_f16(false, A.h, false, Bf[1][kc], (short)0, c1, false, false);
                }
                int rowb = m * 16 + hs * 8;
                #pragma unroll
                for (int r = 0; r < 8; ++r) {
                    Cout[wv][rowb + r][col0] = c0[r];
                    Cout[wv][rowb + r][col1] = c1[r];
                }
            }

            // -------- SH dot + clip --------
            if (active) {
                const float* cp = &Cout[wv][lane][0];
                float r3 = 0.f, g3 = 0.f, b3 = 0.f;
                #pragma unroll
                for (int s = 0; s < 9; ++s) {
                    r3 = fmaf(cp[s],      sh[s], r3);
                    g3 = fmaf(cp[9 + s],  sh[s], g3);
                    b3 = fmaf(cp[18 + s], sh[s], b3);
                }
                cr = fminf(fmaxf(r3 + 0.5f, 0.f), 1.f);
                cg = fminf(fmaxf(g3 + 0.5f, 0.f), 1.f);
                cb = fminf(fmaxf(b3 + 0.5f, 0.f), 1.f);
            }
        }
        racc = fmaf(w, cr, racc);
        gacc = fmaf(w, cg, gacc);
        bacc = fmaf(w, cb, bacc);
    }

    // -------- wave reduction, lane 0 writes --------
    #pragma unroll
    for (int o2 = 16; o2; o2 >>= 1) {
        racc += __shfl_down(racc, o2);
        gacc += __shfl_down(gacc, o2);
        bacc += __shfl_down(bacc, o2);
        wsum += __shfl_down(wsum, o2);
        dsum += __shfl_down(dsum, o2);
    }
    if (lane == 0) {
        float bg = 1.f - wsum;
        out[ray * 3 + 0] = fminf(fmaxf(racc + bg, 0.f), 1.f);
        out[ray * 3 + 1] = fminf(fmaxf(gacc + bg, 0.f), 1.f);
        out[ray * 3 + 2] = fminf(fmaxf(bacc + bg, 0.f), 1.f);
        out[NRAYS * 3 + ray] = dsum;
    }
}

extern "C" void kernel_launch(void* const* d_in, const int* in_sizes, int n_in,
                              void* d_out, int out_size, void* d_ws, size_t ws_size,
                              hipStream_t stream)
{
    (void)in_sizes; (void)n_in; (void)out_size;
    const float* ro = (const float*)d_in[0];
    const float* rd = (const float*)d_in[1];
    const float* dp = (const float*)d_in[2];
    const float* dl = (const float*)d_in[3];
    const float* rp = (const float*)d_in[4];
    const float* rl = (const float*)d_in[5];
    const float* wb = (const float*)d_in[6];
    float* out = (float*)d_out;

    const size_t e_dp = (size_t)3 * NSIG * PHW;
    const size_t e_rp = (size_t)3 * NRGB * PHW;
    const size_t e_dl = (size_t)3 * NSIG * RESI;
    const size_t e_rl = (size_t)3 * NRGB * RESI;
    const size_t need = (e_dp + e_rp + e_dl + e_rl) * sizeof(float);

    if (ws_size >= need) {
        float* t_dp = (float*)d_ws;
        float* t_rp = t_dp + e_dp;
        float* t_dl = t_rp + e_rp;
        float* t_rl = t_dl + e_dl;
        k_tp<NSIG><<<dim3((PHW + 63) / 64, 3), 256, 0, stream>>>(dp, t_dp, PHW);
        k_tp<NRGB><<<dim3((PHW + 63) / 64, 3), 256, 0, stream>>>(rp, t_rp, PHW);
        k_tp<NSIG><<<dim3((RESI + 63) / 64, 3), 256, 0, stream>>>(dl, t_dl, RESI);
        k_tp<NRGB><<<dim3((RESI + 63) / 64, 3), 256, 0, stream>>>(rl, t_rl, RESI);
        k_render<true><<<NRAYS / 4, 128, 0, stream>>>(ro, rd, t_dp, t_dl, t_rp, t_rl, wb, out);
    } else {
        k_render<false><<<NRAYS / 4, 128, 0, stream>>>(ro, rd, dp, dl, rp, rl, wb, out);
    }
}
